// LSTMCell_20641612825104
// MI455X (gfx1250) — matrix operand order repaired and checked by reading, not compiled, not run on hardware
//
#include <hip/hip_runtime.h>

// ---------------------------------------------------------------------------
// Fused LSTM cell for MI455X (gfx1250, wave32, WMMA), two-phase:
//   Phase 1 (prepass): x,h -> bf16 ; Wx,Wh -> bf16 transposed [g][n][k]
//   Phase 2 (fused GEMM): g[4] = x@Wx[g] + h@Wh[g] via v_wmma_f32_16x16x32_bf16
//            with async global->LDS staging (ASYNCcnt) when available,
//            then gate nonlinearities, h_t / c_t written fp32.
// ---------------------------------------------------------------------------

#if defined(__has_builtin)
#  if __has_builtin(__builtin_amdgcn_global_load_async_to_lds_b128) && \
      __has_builtin(__builtin_amdgcn_s_wait_asynccnt)
#    define HAVE_ASYNC 1
#  endif
#endif
#ifndef HAVE_ASYNC
#  define HAVE_ASYNC 0
#endif

typedef __attribute__((ext_vector_type(16))) __bf16 v16bf;
typedef __attribute__((ext_vector_type(8)))  float  v8f;
typedef __attribute__((ext_vector_type(4)))  float  v4f;
typedef __attribute__((ext_vector_type(4)))  int    v4i;
typedef __attribute__((ext_vector_type(8)))  int    v8i;
typedef __attribute__((ext_vector_type(2)))  unsigned v2u;

#define DIM   1024
#define BATCH 8192
#define TM    128   // rows per workgroup
#define TN    64    // cols per workgroup
#define TK    32    // k per step (bf16 WMMA K)

__device__ __forceinline__ unsigned short f2bf(float f) {
  unsigned u = __builtin_bit_cast(unsigned, f);
  u += 0x7FFFu + ((u >> 16) & 1u);          // round-to-nearest-even
  return (unsigned short)(u >> 16);
}
__device__ __forceinline__ unsigned pack2(float a, float b) {
  return (unsigned)f2bf(a) | ((unsigned)f2bf(b) << 16);
}
// two 16B LDS loads -> one v16bf fragment (ds_load_b128 x2)
__device__ __forceinline__ v16bf ldfrag(const unsigned short* p0,
                                        const unsigned short* p1) {
  v4i lo = *(const v4i*)p0;
  v4i hi = *(const v4i*)p1;
  v8i w = { lo.x, lo.y, lo.z, lo.w, hi.x, hi.y, hi.z, hi.w };
  return __builtin_bit_cast(v16bf, w);
}
__device__ __forceinline__ float sigmoidf_fast(float x) {
  return 1.0f / (1.0f + __expf(-x));
}
__device__ __forceinline__ float tanhf_fast(float x) {
  float e2 = __expf(-2.0f * x);
  return (1.0f - e2) / (1.0f + e2);
}

#if HAVE_ASYNC
// one async 16-byte global->LDS copy per lane (global_load_async_to_lds_b128)
__device__ __forceinline__ void async_cp16(const unsigned short* g,
                                           unsigned short* l) {
  __builtin_amdgcn_global_load_async_to_lds_b128(
      (__attribute__((address_space(1))) v4i*)g,
      (__attribute__((address_space(3))) v4i*)l, 0, 0);
}
#endif

// ---------------- prepass: elementwise fp32 -> bf16 ----------------
__global__ __launch_bounds__(256)
void cvt_bf16(const v4f* __restrict__ src, v2u* __restrict__ dst, int n4) {
  int i = blockIdx.x * 256 + threadIdx.x;
  if (i < n4) {
    v4f f = src[i];
    v2u o = { pack2(f.x, f.y), pack2(f.z, f.w) };
    dst[i] = o;
  }
}

// ---------------- prepass: [g][k][n] fp32 -> [g][n][k] bf16 ----------------
__global__ __launch_bounds__(256)
void transpose_bf16(const float* __restrict__ src, unsigned short* __restrict__ dst) {
  __shared__ float tile[32][33];
  const int g  = blockIdx.z;
  const int k0 = blockIdx.x * 32;
  const int n0 = blockIdx.y * 32;
  const int tx = threadIdx.x & 31;
  const int ty = threadIdx.x >> 5;                     // 0..7
  const float* s = src + ((size_t)g * DIM + k0) * DIM + n0;
#pragma unroll
  for (int j = 0; j < 4; ++j)
    tile[ty + j * 8][tx] = s[(size_t)(ty + j * 8) * DIM + tx];
  __syncthreads();
  unsigned short* d = dst + ((size_t)g * DIM + n0) * DIM + k0;
#pragma unroll
  for (int j = 0; j < 2; ++j) {
    int idx = threadIdx.x + j * 256;                   // 512 u32 outputs
    int n   = idx >> 4;
    int kp  = (idx & 15) * 2;
    *(unsigned*)&d[(size_t)n * DIM + kp] = pack2(tile[kp][n], tile[kp + 1][n]);
  }
}

// ---------------- main fused GEMM + gates ----------------
__global__ __launch_bounds__(256)
void lstm_cell_wmma(const unsigned short* __restrict__ Xb,
                    const unsigned short* __restrict__ Hb,
                    const unsigned short* __restrict__ Wb,  // [8][D][D] bf16, [mat][n][k]
                    const float* __restrict__ Cin,
                    float* __restrict__ Out) {
#if HAVE_ASYNC
  __shared__ __align__(16) unsigned short sA[2][2][TM * TK];   // 32 KB
  __shared__ __align__(16) unsigned short sB[2][8][TN * TK];   // 64 KB
#else
  __shared__ __align__(16) unsigned short sA[2][TM * TK];      // 16 KB
  __shared__ __align__(16) unsigned short sB[8][TN * TK];      // 32 KB
#endif

  const int t    = threadIdx.x;
  const int lane = t & 31;
  const int wave = t >> 5;
  const int wm   = wave & 3;
  const int wn   = wave >> 2;
  const int mbase = blockIdx.y * TM;
  const int nbase = blockIdx.x * TN;

  v8f acc[4][2][2] = {};                  // [gate][mt][nt]

  // staging jobs: A -> (row, 32B half); B -> (matrix, 2 n-rows)
  const int arow = t >> 1;
  const int akh  = (t & 1) * 16;          // bf16 units
  const int bmat = t >> 5;
  const int brow = (t & 31) * 2;

  const unsigned short* gxp =
      Xb + (size_t)(mbase + arow) * DIM + akh;
  const unsigned short* ghp =
      Hb + (size_t)(mbase + arow) * DIM + akh;
  const unsigned short* wp0 =
      Wb + (size_t)bmat * DIM * DIM + (size_t)(nbase + brow) * DIM;

  const int lrow  = lane & 15;
  const int ahalf = (lane < 16) ? 0 : 16;   // A: K-half base per lane
  const int bko   = (lane < 16) ? 0 : 16;   // B: K-half base per lane

  // 32 WMMAs on the current LDS tile, B fragments double-buffered in regs
  auto do_compute = [&](const unsigned short (*cA)[TM * TK],
                        const unsigned short (*cB)[TN * TK]) {
    v16bf aX[2], aH[2];
#pragma unroll
    for (int mt = 0; mt < 2; ++mt) {
      const unsigned short* ax = &cA[0][(wm * 32 + mt * 16 + lrow) * TK];
      const unsigned short* ah = &cA[1][(wm * 32 + mt * 16 + lrow) * TK];
      aX[mt] = ldfrag(ax + ahalf, ax + ahalf + 8);
      aH[mt] = ldfrag(ah + ahalf, ah + ahalf + 8);
    }
    auto bp = [&](int mg, int nt) {
      return &cB[mg][(wn * 32 + nt * 16 + lrow) * TK + bko];
    };
    v16bf f0 = ldfrag(bp(0, 0), bp(0, 0) + 8);
    v16bf f1 = ldfrag(bp(0, 1), bp(0, 1) + 8);
#pragma unroll
    for (int mg = 0; mg < 8; ++mg) {
      v16bf n0 = f0, n1 = f1;
      if (mg < 7) {                          // lookahead fragments
        n0 = ldfrag(bp(mg + 1, 0), bp(mg + 1, 0) + 8);
        n1 = ldfrag(bp(mg + 1, 1), bp(mg + 1, 1) + 8);
      }
      const v16bf* a = (mg < 4) ? aX : aH;
      const int g = mg & 3;
      acc[g][0][0] = __builtin_amdgcn_wmma_f32_16x16x32_bf16(
          false, a[0], false, f0, (short)0, acc[g][0][0], false, false);
      acc[g][1][0] = __builtin_amdgcn_wmma_f32_16x16x32_bf16(
          false, a[1], false, f0, (short)0, acc[g][1][0], false, false);
      acc[g][0][1] = __builtin_amdgcn_wmma_f32_16x16x32_bf16(
          false, a[0], false, f1, (short)0, acc[g][0][1], false, false);
      acc[g][1][1] = __builtin_amdgcn_wmma_f32_16x16x32_bf16(
          false, a[1], false, f1, (short)0, acc[g][1][1], false, false);
      f0 = n0; f1 = n1;
    }
  };

#if HAVE_ASYNC
  // ---- async double-buffered staging: global -> LDS, no VGPR round-trip ----
  auto issue_tile = [&](int k0, int p) {
    async_cp16(gxp + k0,      &sA[p][0][arow * TK + akh]);
    async_cp16(gxp + k0 + 8,  &sA[p][0][arow * TK + akh + 8]);
    async_cp16(ghp + k0,      &sA[p][1][arow * TK + akh]);
    async_cp16(ghp + k0 + 8,  &sA[p][1][arow * TK + akh + 8]);
    const unsigned short* w0 = wp0 + k0;
    unsigned short* d0 = &sB[p][bmat][brow * TK];
#pragma unroll
    for (int j = 0; j < 4; ++j) async_cp16(w0 + j * 8, d0 + j * 8);
    const unsigned short* w1 = wp0 + DIM + k0;
    unsigned short* d1 = &sB[p][bmat][(brow + 1) * TK];
#pragma unroll
    for (int j = 0; j < 4; ++j) async_cp16(w1 + j * 8, d1 + j * 8);
  };

  issue_tile(0, 0);
  __builtin_amdgcn_s_wait_asynccnt(0);
  __syncthreads();
  int p = 0;
  for (int k0 = 0; k0 < DIM; k0 += TK, p ^= 1) {
    const bool more = (k0 + TK) < DIM;
    if (more) issue_tile(k0 + TK, p ^ 1);   // DMA fills other buffer...
    do_compute(sA[p], sB[p]);               // ...while we run 32 WMMAs
    if (more) {
      __builtin_amdgcn_s_wait_asynccnt(0);
      __syncthreads();
    }
  }
#else
  // ---- fallback: register-pipelined synchronous staging ----
  auto load_tile = [&](int k0, v4i rx[2], v4i rh[2], v4i rw[8]) {
    const v4i* gx = (const v4i*)(gxp + k0);
    rx[0] = gx[0]; rx[1] = gx[1];
    const v4i* gh = (const v4i*)(ghp + k0);
    rh[0] = gh[0]; rh[1] = gh[1];
    const v4i* g0 = (const v4i*)(wp0 + k0);
    const v4i* g1 = (const v4i*)(wp0 + DIM + k0);
    rw[0] = g0[0]; rw[1] = g0[1]; rw[2] = g0[2]; rw[3] = g0[3];
    rw[4] = g1[0]; rw[5] = g1[1]; rw[6] = g1[2]; rw[7] = g1[3];
    if (k0 + TK < DIM)
      __builtin_prefetch(wp0 + k0 + TK, 0, 1);
  };
  auto store_tile = [&](const v4i rx[2], const v4i rh[2], const v4i rw[8]) {
    v4i* dx = (v4i*)&sA[0][arow * TK + akh];
    dx[0] = rx[0]; dx[1] = rx[1];
    v4i* dh = (v4i*)&sA[1][arow * TK + akh];
    dh[0] = rh[0]; dh[1] = rh[1];
    v4i* d0 = (v4i*)&sB[bmat][brow * TK];
    d0[0] = rw[0]; d0[1] = rw[1]; d0[2] = rw[2]; d0[3] = rw[3];
    v4i* d1 = (v4i*)&sB[bmat][(brow + 1) * TK];
    d1[0] = rw[4]; d1[1] = rw[5]; d1[2] = rw[6]; d1[3] = rw[7];
  };

  v4i rx[2], rh[2], rw[8];
  load_tile(0, rx, rh, rw);
  store_tile(rx, rh, rw);
  __syncthreads();
  for (int k0 = 0; k0 < DIM; k0 += TK) {
    const bool more = (k0 + TK) < DIM;
    if (more) load_tile(k0 + TK, rx, rh, rw);
    do_compute(sA, sB);
    if (more) {
      __syncthreads();
      store_tile(rx, rh, rw);
      __syncthreads();
    }
  }
#endif

  // fused gate epilogue
  float* Hout = Out;
  float* Cout = Out + (size_t)BATCH * DIM;
  const int mrow0 = mbase + wm * 32;
  const int ncol0 = nbase + wn * 32;
#pragma unroll
  for (int mt = 0; mt < 2; ++mt) {
#pragma unroll
    for (int nt = 0; nt < 2; ++nt) {
      const int col   = ncol0 + nt * 16 + (lane & 15);
      const int rbase = mrow0 + mt * 16 + ((lane < 16) ? 0 : 8);
#pragma unroll
      for (int v = 0; v < 8; ++v) {
        const int row = rbase + v;
        float ig = sigmoidf_fast(acc[0][mt][nt][v]);
        float fg = sigmoidf_fast(acc[1][mt][nt][v]);
        float og = sigmoidf_fast(acc[2][mt][nt][v]);
        float ug = tanhf_fast(acc[3][mt][nt][v]);
        size_t idx = (size_t)row * DIM + col;
        float ct = ig * ug + fg * Cin[idx];
        Hout[idx] = og * tanhf_fast(ct);
        Cout[idx] = ct;
      }
    }
  }
}

extern "C" void kernel_launch(void* const* d_in, const int* in_sizes, int n_in,
                              void* d_out, int out_size, void* d_ws,
                              size_t ws_size, hipStream_t stream) {
  const float* x  = (const float*)d_in[0];
  const float* h  = (const float*)d_in[1];
  const float* c  = (const float*)d_in[2];
  const float* Wx = (const float*)d_in[3];
  const float* Wh = (const float*)d_in[4];
  float* out = (float*)d_out;

  // workspace layout (bf16): xb | hb | Wb (8 matrices, [mat][n][k])
  unsigned short* xb = (unsigned short*)d_ws;
  unsigned short* hb = xb + (size_t)BATCH * DIM;
  unsigned short* wb = hb + (size_t)BATCH * DIM;       // 48 MB total

  const int n4 = BATCH * DIM / 4;
  cvt_bf16<<<dim3((n4 + 255) / 256), 256, 0, stream>>>((const v4f*)x, (v2u*)xb, n4);
  cvt_bf16<<<dim3((n4 + 255) / 256), 256, 0, stream>>>((const v4f*)h, (v2u*)hb, n4);
  transpose_bf16<<<dim3(32, 32, 4), 256, 0, stream>>>(Wx, wb);
  transpose_bf16<<<dim3(32, 32, 4), 256, 0, stream>>>(Wh, wb + (size_t)4 * DIM * DIM);

  lstm_cell_wmma<<<dim3(DIM / TN, BATCH / TM), 256, 0, stream>>>(
      xb, hb, wb, c, out);
}
